// PerformerAttention_17660905521521
// MI455X (gfx1250) — matrix operand order repaired and checked by reading, not compiled, not run on hardware
//
#include <hip/hip_runtime.h>

#define BQ 16
#define NT 577
#define CC 768
#define HH 12
#define DD 64
#define MM (BQ * NT)     /* 9232 rows (b,n) */
#define KK CC            /* 768  */
#define NFQ (3 * CC)     /* 2304 */

typedef __attribute__((ext_vector_type(16))) __bf16 v16bf;
typedef __attribute__((ext_vector_type(8)))  float  v8f;
typedef __attribute__((ext_vector_type(4)))  unsigned int v4u;

#define WMMA_BF16(a, b, c) \
  __builtin_amdgcn_wmma_f32_16x16x32_bf16(false, (a), false, (b), (short)0, (c), false, false)

__device__ __forceinline__ unsigned short bf16r(float x) {
  unsigned int u = __float_as_uint(x);
  u += 0x7fffu + ((u >> 16) & 1u);          // round-to-nearest-even
  return (unsigned short)(u >> 16);
}
__device__ __forceinline__ float bf2f(unsigned int bits16) {
  return __uint_as_float(bits16 << 16);
}

// A-fragment (16x32 bf16, ISA 7.12.2): lane row = lane&15.
// Caller passes p = base + row*ld + (lane>>4)*8; K chunks at +0 and +16.
__device__ __forceinline__ v16bf ldfragA(const unsigned short* p) {
  union { v4u u[2]; v16bf v; } t;
  t.u[0] = *(const v4u*)(p);
  t.u[1] = *(const v4u*)(p + 16);
  return t.v;
}

// B-fragment (32x16 bf16, ISA 7.12.4 column layout): lane col = lane&15,
// lanes 0-15 hold K=0..7,16..23, lanes 16-31 hold K=8..15,24..31.
// Caller passes p = base + col*ld + (lane>>4)*8; chunks at +0 and +16.
__device__ __forceinline__ v16bf ldfragB(const unsigned short* p) {
  union { v4u u[2]; v16bf v; } t;
  t.u[0] = *(const v4u*)(p);
  t.u[1] = *(const v4u*)(p + 16);
  return t.v;
}

// ---------------------------------------------------------------- convert ---
__global__ __launch_bounds__(256) void vit_cvt_bf16(const float* __restrict__ in,
                                                    unsigned short* __restrict__ out,
                                                    int n4) {
  int i = blockIdx.x * 256 + threadIdx.x;
  if (i >= n4) return;
  float4 f = ((const float4*)in)[i];
  union { unsigned short s[4]; unsigned long long u; } r;
  r.s[0] = bf16r(f.x); r.s[1] = bf16r(f.y); r.s[2] = bf16r(f.z); r.s[3] = bf16r(f.w);
  ((unsigned long long*)out)[i] = r.u;
}

// ------------------------------------------------------------- QKV GEMM -----
// qkv = x @ wqkv^T, scattered into q/k/v [B*H, N, D] bf16 buffers.
__global__ __launch_bounds__(256) void vit_gemm_qkv(const unsigned short* __restrict__ A,
                                                    const unsigned short* __restrict__ W,
                                                    unsigned short* __restrict__ Qb,
                                                    unsigned short* __restrict__ Kb,
                                                    unsigned short* __restrict__ Vb) {
  const int lane = threadIdx.x & 31;
  const int wave = threadIdx.x >> 5;
  const int l15 = lane & 15, half = lane >> 4;
  const int m_base = blockIdx.y * 128 + (wave >> 1) * 32;
  const int n_base = blockIdx.x * 128 + (wave & 1) * 64;

  const unsigned short* aptr[2];
  const unsigned short* bptr[4];
#pragma unroll
  for (int i = 0; i < 2; ++i) {
    int r = m_base + 16 * i + l15; if (r > MM - 1) r = MM - 1;
    aptr[i] = A + (size_t)r * KK + half * 8;
  }
#pragma unroll
  for (int j = 0; j < 4; ++j)
    bptr[j] = W + (size_t)(n_base + 16 * j + l15) * KK + half * 8;

  v8f zero = {};
  v8f acc[2][4];
#pragma unroll
  for (int i = 0; i < 2; ++i)
#pragma unroll
    for (int j = 0; j < 4; ++j) acc[i][j] = zero;

  for (int kk = 0; kk < KK; kk += 32) {
    v16bf af[2], bf[4];
#pragma unroll
    for (int i = 0; i < 2; ++i) af[i] = ldfragA(aptr[i] + kk);
#pragma unroll
    for (int j = 0; j < 4; ++j) bf[j] = ldfragB(bptr[j] + kk);
#pragma unroll
    for (int i = 0; i < 2; ++i)
#pragma unroll
      for (int j = 0; j < 4; ++j) acc[i][j] = WMMA_BF16(af[i], bf[j], acc[i][j]);
  }

#pragma unroll
  for (int i = 0; i < 2; ++i) {
#pragma unroll
    for (int rr = 0; rr < 8; ++rr) {
      int row = m_base + 16 * i + rr + 8 * half;
      if (row >= MM) continue;
      int b = row / NT, tok = row % NT;
#pragma unroll
      for (int j = 0; j < 4; ++j) {
        int f = n_base + 16 * j + l15;
        int part = f / CC, wi = f % CC;
        int h = wi >> 6, d = wi & 63;
        unsigned short* dst = (part == 0) ? Qb : (part == 1) ? Kb : Vb;
        dst[((size_t)(b * HH + h) * NT + tok) * DD + d] = bf16r(acc[i][j][rr]);
      }
    }
  }
}

// ----------------------------------------------------------------- RoPE -----
__global__ __launch_bounds__(256) void vit_rope(unsigned short* __restrict__ Qb,
                                                unsigned short* __restrict__ Kb,
                                                const float* __restrict__ fc,
                                                const float* __restrict__ fs) {
  const int total = BQ * HH * (NT - 1) * (DD / 2);
  int idx = blockIdx.x * 256 + threadIdx.x;
  if (idx >= total) return;
  int p = idx & 31;
  int t = (idx >> 5) % (NT - 1);
  int bh = idx / ((NT - 1) * 32);
  float c = fc[t * 32 + p], s = fs[t * 32 + p];
  size_t base = ((size_t)bh * NT + (t + 1)) * DD + 2 * p;
  {
    unsigned int u = *(unsigned int*)(Qb + base);
    float te = bf2f(u & 0xffffu), to = bf2f(u >> 16);
    *(unsigned int*)(Qb + base) =
        (unsigned int)bf16r(te * c - to * s) | ((unsigned int)bf16r(te * s + to * c) << 16);
  }
  {
    unsigned int u = *(unsigned int*)(Kb + base);
    float te = bf2f(u & 0xffffu), to = bf2f(u >> 16);
    *(unsigned int*)(Kb + base) =
        (unsigned int)bf16r(te * c - to * s) | ((unsigned int)bf16r(te * s + to * c) << 16);
  }
}

// ---------------------------------------------------- flash attention -------
// One wave handles one 16-row query tile of one (b,h). 4 waves / block.
#define QTILES_PER_BH 37          /* ceil(577/16) */
__global__ __launch_bounds__(128) void vit_attn(const unsigned short* __restrict__ Qb,
                                                const unsigned short* __restrict__ Kb,
                                                const unsigned short* __restrict__ Vb,
                                                unsigned short* __restrict__ Ob) {
  __shared__ unsigned short ldsP[4][16 * 32];   // P tile, row-major 16x32
  __shared__ unsigned short ldsVT[4][64 * 32];  // V^T tile, 64(d) x 32(keys)

  const int lane = threadIdx.x & 31;
  const int wave = threadIdx.x >> 5;
  const int l15 = lane & 15, half = lane >> 4;
  const int tile = blockIdx.x * 4 + wave;
  const int bh = tile / QTILES_PER_BH;
  const int mt = tile % QTILES_PER_BH;
  const int b = bh / HH, h = bh % HH;

  const unsigned short* q = Qb + (size_t)bh * NT * DD;
  const unsigned short* k = Kb + (size_t)bh * NT * DD;
  const unsigned short* v = Vb + (size_t)bh * NT * DD;
  unsigned short* Pl = ldsP[wave];
  unsigned short* VT = ldsVT[wave];

  int qrow = mt * 16 + l15; if (qrow > NT - 1) qrow = NT - 1;
  v16bf qf0 = ldfragA(q + (size_t)qrow * DD + half * 8);
  v16bf qf1 = ldfragA(q + (size_t)qrow * DD + 32 + half * 8);

  v8f zero = {};
  v8f acc[4] = {zero, zero, zero, zero};
  float mrow[8], lrow[8];
#pragma unroll
  for (int r = 0; r < 8; ++r) { mrow[r] = -3.0e38f; lrow[r] = 0.0f; }

  const float scale = 0.125f;                    // 1/sqrt(64)

  for (int j0 = 0; j0 < NT; j0 += 32) {          // 19 iterations, uniform
    // ---- S = Q @ K^T for 32 keys (two 16x16 tiles) ----
    v8f s[2];
#pragma unroll
    for (int t = 0; t < 2; ++t) {
      int krow = j0 + 16 * t + l15;
      int kr = (krow < NT) ? krow : 0;
      // B operand: column = key (lane&15); K dim = D, table-order halves.
      v16bf b0 = ldfragB(k + (size_t)kr * DD + half * 8);        // d = 0..31
      v16bf b1 = ldfragB(k + (size_t)kr * DD + 32 + half * 8);   // d = 32..63
      s[t] = WMMA_BF16(qf0, b0, zero);
      s[t] = WMMA_BF16(qf1, b1, s[t]);
    }
    float sv[2][8];
#pragma unroll
    for (int t = 0; t < 2; ++t) {
      bool valid = (j0 + 16 * t + l15) < NT;
#pragma unroll
      for (int r = 0; r < 8; ++r) sv[t][r] = valid ? s[t][r] * scale : -1.0e30f;
    }
    // ---- online softmax: row max over 32 cols (16-lane halves hold rows) ----
    float mx[8];
#pragma unroll
    for (int r = 0; r < 8; ++r) mx[r] = fmaxf(sv[0][r], sv[1][r]);
#pragma unroll
    for (int off = 8; off >= 1; off >>= 1)
#pragma unroll
      for (int r = 0; r < 8; ++r) mx[r] = fmaxf(mx[r], __shfl_xor(mx[r], off, 32));
    float sf[8];
#pragma unroll
    for (int r = 0; r < 8; ++r) {
      float mnew = fmaxf(mrow[r], mx[r]);
      sf[r] = __expf(mrow[r] - mnew);
      mrow[r] = mnew;
    }
#pragma unroll
    for (int c = 0; c < 4; ++c)
#pragma unroll
      for (int r = 0; r < 8; ++r) acc[c][r] *= sf[r];
    // ---- P = exp(S-m), stage to LDS, accumulate row sums ----
    float rs[8];
#pragma unroll
    for (int r = 0; r < 8; ++r) rs[r] = 0.0f;
#pragma unroll
    for (int t = 0; t < 2; ++t)
#pragma unroll
      for (int r = 0; r < 8; ++r) {
        float p = __expf(sv[t][r] - mrow[r]);
        rs[r] += p;
        Pl[(r + 8 * half) * 32 + 16 * t + l15] = bf16r(p);
      }
#pragma unroll
    for (int off = 8; off >= 1; off >>= 1)
#pragma unroll
      for (int r = 0; r < 8; ++r) rs[r] += __shfl_xor(rs[r], off, 32);
#pragma unroll
    for (int r = 0; r < 8; ++r) lrow[r] = lrow[r] * sf[r] + rs[r];

    // ---- stage V^T (64 d x 32 keys): lane owns d pair (2*lane, 2*lane+1) ----
#pragma unroll
    for (int g = 0; g < 4; ++g) {
      union { unsigned short s[8]; v4u u; } r0, r1;
#pragma unroll
      for (int e = 0; e < 8; ++e) {
        int key = j0 + g * 8 + e;
        int kc = (key < NT) ? key : 0;
        unsigned int w = *(const unsigned int*)(v + (size_t)kc * DD + 2 * lane);
        r0.s[e] = (unsigned short)(w & 0xffffu);
        r1.s[e] = (unsigned short)(w >> 16);
      }
      *(v4u*)&VT[(2 * lane) * 32 + g * 8]     = r0.u;
      *(v4u*)&VT[(2 * lane + 1) * 32 + g * 8] = r1.u;
    }
    __syncthreads();

    // ---- O += P @ V  (A = P 16x32 from LDS, B = V via V^T rows in LDS) ----
    v16bf pf = ldfragA(Pl + l15 * 32 + half * 8);
#pragma unroll
    for (int c = 0; c < 4; ++c) {
      // B operand: column = d (lane&15) within chunk c; K = 32 keys in table order.
      v16bf vf = ldfragB(VT + (16 * c + l15) * 32 + half * 8);
      acc[c] = WMMA_BF16(pf, vf, acc[c]);
    }
    __syncthreads();
  }

  // ---- normalize and store [B,N,C] bf16 ----
#pragma unroll
  for (int r = 0; r < 8; ++r) {
    int row = mt * 16 + r + 8 * half;
    if (row >= NT) continue;
    float inv = 1.0f / lrow[r];
#pragma unroll
    for (int c = 0; c < 4; ++c)
      Ob[((size_t)(b * NT + row)) * CC + h * DD + 16 * c + l15] = bf16r(acc[c][r] * inv);
  }
}

// ------------------------------------------------------------ proj GEMM -----
__global__ __launch_bounds__(256) void vit_gemm_proj(const unsigned short* __restrict__ A,
                                                     const unsigned short* __restrict__ W,
                                                     const float* __restrict__ bias,
                                                     float* __restrict__ out) {
  const int lane = threadIdx.x & 31;
  const int wave = threadIdx.x >> 5;
  const int l15 = lane & 15, half = lane >> 4;
  const int m_base = blockIdx.y * 128 + (wave >> 1) * 32;
  const int n_base = blockIdx.x * 128 + (wave & 1) * 64;

  const unsigned short* aptr[2];
  const unsigned short* bptr[4];
#pragma unroll
  for (int i = 0; i < 2; ++i) {
    int r = m_base + 16 * i + l15; if (r > MM - 1) r = MM - 1;
    aptr[i] = A + (size_t)r * KK + half * 8;
  }
#pragma unroll
  for (int j = 0; j < 4; ++j)
    bptr[j] = W + (size_t)(n_base + 16 * j + l15) * KK + half * 8;

  v8f zero = {};
  v8f acc[2][4];
#pragma unroll
  for (int i = 0; i < 2; ++i)
#pragma unroll
    for (int j = 0; j < 4; ++j) acc[i][j] = zero;

  for (int kk = 0; kk < KK; kk += 32) {
    v16bf af[2], bf[4];
#pragma unroll
    for (int i = 0; i < 2; ++i) af[i] = ldfragA(aptr[i] + kk);
#pragma unroll
    for (int j = 0; j < 4; ++j) bf[j] = ldfragB(bptr[j] + kk);
#pragma unroll
    for (int i = 0; i < 2; ++i)
#pragma unroll
      for (int j = 0; j < 4; ++j) acc[i][j] = WMMA_BF16(af[i], bf[j], acc[i][j]);
  }

#pragma unroll
  for (int i = 0; i < 2; ++i) {
#pragma unroll
    for (int rr = 0; rr < 8; ++rr) {
      int row = m_base + 16 * i + rr + 8 * half;
      if (row >= MM) continue;
#pragma unroll
      for (int j = 0; j < 4; ++j) {
        int f = n_base + 16 * j + l15;
        out[(size_t)row * CC + f] = acc[i][j][rr] + bias[f];
      }
    }
  }
}

// ---------------------------------------------------------------- launch ----
extern "C" void kernel_launch(void* const* d_in, const int* in_sizes, int n_in,
                              void* d_out, int out_size, void* d_ws, size_t ws_size,
                              hipStream_t stream) {
  const float* x     = (const float*)d_in[0];   // [B,N,C]
  const float* wqkv  = (const float*)d_in[1];   // [3C,C]
  const float* wproj = (const float*)d_in[2];   // [C,C]
  const float* bproj = (const float*)d_in[3];   // [C]
  const float* fcos  = (const float*)d_in[4];   // [N-1,D/2]
  const float* fsin  = (const float*)d_in[5];   // [N-1,D/2]
  float* out = (float*)d_out;

  char* w = (char*)d_ws;
  size_t off = 0;
  auto carve = [&](size_t elems) {
    char* p = w + off;
    off = (off + elems * 2 + 255) & ~(size_t)255;
    return (unsigned short*)p;
  };
  unsigned short* xb     = carve((size_t)MM * KK);
  unsigned short* wqkvb  = carve((size_t)NFQ * KK);
  unsigned short* wprojb = carve((size_t)CC * KK);
  unsigned short* qb     = carve((size_t)BQ * HH * NT * DD);
  unsigned short* kb     = carve((size_t)BQ * HH * NT * DD);
  unsigned short* vb     = carve((size_t)BQ * HH * NT * DD);
  unsigned short* ob     = carve((size_t)MM * CC);
  (void)ws_size; (void)in_sizes; (void)n_in; (void)out_size;

  {
    int n4 = (MM * KK) / 4;
    vit_cvt_bf16<<<(n4 + 255) / 256, 256, 0, stream>>>(x, xb, n4);
  }
  {
    int n4 = (NFQ * KK) / 4;
    vit_cvt_bf16<<<(n4 + 255) / 256, 256, 0, stream>>>(wqkv, wqkvb, n4);
  }
  {
    int n4 = (CC * KK) / 4;
    vit_cvt_bf16<<<(n4 + 255) / 256, 256, 0, stream>>>(wproj, wprojb, n4);
  }
  vit_gemm_qkv<<<dim3(NFQ / 128, (MM + 127) / 128), 256, 0, stream>>>(xb, wqkvb, qb, kb, vb);
  {
    int total = BQ * HH * (NT - 1) * (DD / 2);
    vit_rope<<<(total + 255) / 256, 256, 0, stream>>>(qb, kb, fcos, fsin);
  }
  {
    int tiles = BQ * HH * QTILES_PER_BH;      // 7104, divisible by 4
    vit_attn<<<tiles / 4, 128, 0, stream>>>(qb, kb, vb, ob);
  }
  vit_gemm_proj<<<dim3(CC / 128, (MM + 127) / 128), 256, 0, stream>>>(ob, wprojb, bproj, out);
}
